// MultimodalLSTM_29317446763043
// MI455X (gfx1250) — matrix operand order repaired and checked by reading, not compiled, not run on hardware
//
#include <hip/hip_runtime.h>
#include <hip/hip_bf16.h>

// ---------------------------------------------------------------------------
// Fused 2-layer LSTM + FC head for MI455X (gfx1250), wave32 + WMMA bf16 + TDM.
//
// x[256,512,256] f32 -> LSTM(H=64) -> LSTM(H=64) -> sigmoid(fc) -> [256,1]
// HBM floor: 134 MB read of x @ 23.3 TB/s ~ 5.8 us -> read x exactly once,
// keep all weights on-chip, hide the serial scan behind the x stream.
//
// 16 workgroups x 16 batch rows (WMMA M=16), 8 waves. Each wave owns 2 of the
// 16 gate-column tiles. Whh0/Wih1/Whh1 live in LDS (bf16); Wih0 (128 KB bf16,
// pre-converted into d_ws) streams from L2 and is hoisted into VGPRs by the
// compiler (loop-invariant B-fragments).
//
// x staging uses the Tensor Data Mover: one tensor_load_to_lds per timestep
// (16 rows x 1 KB, row stride T*I*4 B), double-buffered so the DMA for t+1
// overlaps the WMMA work of t; completion via s_wait_tensorcnt 0.
// ---------------------------------------------------------------------------

typedef __attribute__((ext_vector_type(16))) __bf16 v16bf;
typedef __attribute__((ext_vector_type(8)))  float  v8f;
typedef __attribute__((ext_vector_type(4)))  unsigned int u32x4;
typedef __attribute__((ext_vector_type(8)))  int    i32x8;
typedef __attribute__((ext_vector_type(4)))  int    i32x4;

#define ISZ   256   // input size
#define HSZ   64    // hidden
#define GSZ   256   // 4*H gate width
#define TSZ   512   // timesteps
#define BSZ   256   // batch
#define BTILE 16    // batch rows per workgroup (WMMA M)
#define NTHREADS 256

#if __has_builtin(__builtin_amdgcn_tensor_load_to_lds) && \
    __has_builtin(__builtin_amdgcn_s_wait_tensorcnt)
#define USE_TDM 1
#else
#define USE_TDM 0
#endif

union FragU { v16bf v; uint4 q[2]; };

// A-fragment: 16x32 bf16 tile (ISA 7.12.2 16-bit A layout).
// lane<16: K in {0..7, 16..23}; lane>=16: K in {8..15, 24..31}.
__device__ __forceinline__ v16bf load_frag_a(const __bf16* base, int rowStride,
                                             int m, int k0, int lane) {
    const int o1 = (lane < 16) ? 0 : 8;
    const __bf16* p = base + m * rowStride + k0 + o1;
    FragU f;
    f.q[0] = *(const uint4*)(p);
    f.q[1] = *(const uint4*)(p + 16);
    return f.v;
}

// B-fragment: 32x16 (KxN). Weights stored row-major [gate][k], same K order as A.
// lane<16: K in {0..7, 16..23}; lane>=16: K in {8..15, 24..31}.
__device__ __forceinline__ v16bf load_frag_b(const __bf16* base, int rowStride,
                                             int row, int k0, int lane) {
    const int koff = (lane < 16) ? 0 : 8;
    const __bf16* p = base + row * rowStride + k0 + koff;
    FragU f;
    f.q[0] = *(const uint4*)(p);
    f.q[1] = *(const uint4*)(p + 16);
    return f.v;
}

__device__ __forceinline__ float sigmoidf_(float x) {
    return 1.0f / (1.0f + __expf(-x));
}
__device__ __forceinline__ float tanhf_(float x) {
    return 2.0f / (1.0f + __expf(-2.0f * x)) - 1.0f;
}

#if USE_TDM
// Issue a TDM load of a 16x256 f32 tile (row stride T*I elements) into LDS.
// D# per CDNA5 ISA 8.3-8.6; groups 2/3 zero (2-D tensor), no pad/iterate.
__device__ __forceinline__ void tdm_load_x_tile(const float* gsrc,
                                                unsigned lds_off) {
    const unsigned long long ga = (unsigned long long)(uintptr_t)gsrc;
    u32x4 g0;
    g0[0] = 1u;                                   // count=1, user descriptor
    g0[1] = lds_off;                              // lds_addr (bytes)
    g0[2] = (unsigned)(ga & 0xffffffffu);         // global_addr[31:0]
    g0[3] = (unsigned)((ga >> 32) & 0x01ffffffu)  // global_addr[56:32]
          | (2u << 30);                           // type=2 ("image")
    i32x8 g1;
    g1[0] = (int)(2u << 16);          // data_size=4B, workgroup_mask=0
    g1[1] = (int)(256u << 16);        // tensor_dim0 = 256 (bits 79:48 lo)
    g1[2] = (int)(16u << 16);         // tensor_dim1 = 16  (bits 111:80 lo)
    g1[3] = (int)(256u << 16);        // tile_dim0 = 256   (bits 127:112)
    g1[4] = 16;                       // tile_dim1 = 16, tile_dim2 = 0
    g1[5] = (int)(TSZ * ISZ);         // tensor_dim0_stride = 131072 elements
    g1[6] = 0;                        // stride hi / tensor_dim1_stride lo
    g1[7] = 0;
    const i32x4 z4 = {0, 0, 0, 0};
#if defined(__clang_major__) && (__clang_major__ >= 23)
    const i32x8 z8 = {0, 0, 0, 0, 0, 0, 0, 0};
    __builtin_amdgcn_tensor_load_to_lds(g0, g1, z4, z4, z8, 0);
#else
    __builtin_amdgcn_tensor_load_to_lds(g0, g1, z4, z4, 0);
#endif
}
#endif

__global__ void convert_wih0_kernel(const float* __restrict__ w,
                                    __bf16* __restrict__ o, int n) {
    int i = blockIdx.x * blockDim.x + threadIdx.x;
    if (i < n) o[i] = (__bf16)w[i];
}

__global__ __launch_bounds__(NTHREADS)
void lstm_fused_kernel(const float* __restrict__ x,
                       const __bf16* __restrict__ wih0bf,   // [256][256] bf16
                       const float* __restrict__ whh0,      // [256][64]
                       const float* __restrict__ bih0,
                       const float* __restrict__ bhh0,
                       const float* __restrict__ wih1,      // [256][64]
                       const float* __restrict__ whh1,      // [256][64]
                       const float* __restrict__ bih1,
                       const float* __restrict__ bhh1,
                       const float* __restrict__ wfc,       // [64]
                       const float* __restrict__ bfc,       // [1]
                       float* __restrict__ out) {           // [256]
    __shared__ __bf16 sWhh0[GSZ * HSZ];     // 32 KB  [g][k]
    __shared__ __bf16 sWih1[GSZ * HSZ];     // 32 KB
    __shared__ __bf16 sWhh1[GSZ * HSZ];     // 32 KB
    __shared__ float  sXf[2][BTILE * ISZ];  // 32 KB  TDM landing (f32, 2 bufs)
    __shared__ __bf16 sX [BTILE * ISZ];     // 8 KB   current x tile (bf16)
    __shared__ __bf16 sH0[BTILE * HSZ];     // 2 KB
    __shared__ __bf16 sH1[BTILE * HSZ];     // 2 KB
    __shared__ float  sG [BTILE * GSZ];     // 16 KB  gate pre-activations
    __shared__ float  sB0[GSZ];
    __shared__ float  sB1[GSZ];

    const int tid  = threadIdx.x;
    const int lane = tid & 31;
    const int wv   = tid >> 5;          // 8 waves
    const int b0   = blockIdx.x * BTILE;

    // ---- preload weights/biases into LDS, zero h ----
    for (int i = tid; i < GSZ * HSZ; i += NTHREADS) {
        sWhh0[i] = (__bf16)whh0[i];
        sWih1[i] = (__bf16)wih1[i];
        sWhh1[i] = (__bf16)whh1[i];
    }
    for (int g = tid; g < GSZ; g += NTHREADS) {
        sB0[g] = bih0[g] + bhh0[g];
        sB1[g] = bih1[g] + bhh1[g];
    }
    for (int i = tid; i < BTILE * HSZ; i += NTHREADS) {
        sH0[i] = (__bf16)0.0f;
        sH1[i] = (__bf16)0.0f;
    }
    float c0r[4] = {0.f, 0.f, 0.f, 0.f};
    float c1r[4] = {0.f, 0.f, 0.f, 0.f};

#if USE_TDM
    // Prologue: kick off DMA of the t=0 tile while the preload settles.
    if (wv == 0) {
        tdm_load_x_tile(x + (size_t)b0 * TSZ * ISZ,
                        (unsigned)(uintptr_t)&sXf[0][0]);
    }
#endif
    __syncthreads();

    const int n    = lane & 15;           // WMMA column within tile
    const int moff = (lane < 16) ? 0 : 8; // C/D row offset for this lane half

    for (int t = 0; t < TSZ; ++t) {
        const int buf = t & 1;
#if USE_TDM
        // ---- wait for tile t, then immediately start DMA of tile t+1 ----
        if (wv == 0) __builtin_amdgcn_s_wait_tensorcnt(0);
        __syncthreads();                     // sXf[buf] valid for all waves
        if (wv == 0 && (t + 1) < TSZ) {
            tdm_load_x_tile(x + ((size_t)b0 * TSZ + (t + 1)) * ISZ,
                            (unsigned)(uintptr_t)&sXf[buf ^ 1][0]);
        }
        // ---- LDS f32 -> LDS bf16 convert pass (feeds WMMA A-fragments) ----
        #pragma unroll
        for (int i = 0; i < 4; ++i) {
            const int q = tid + i * NTHREADS;   // 0..1023
            const int m = q >> 6;
            const int k = (q & 63) * 4;
            const float4 v = *(const float4*)&sXf[buf][m * ISZ + k];
            sX[m * ISZ + k + 0] = (__bf16)v.x;
            sX[m * ISZ + k + 1] = (__bf16)v.y;
            sX[m * ISZ + k + 2] = (__bf16)v.z;
            sX[m * ISZ + k + 3] = (__bf16)v.w;
        }
#else
        (void)buf;
        // Fallback: synchronous per-thread staging with prefetch of t+1.
        #pragma unroll
        for (int i = 0; i < 4; ++i) {
            const int q = tid + i * NTHREADS;
            const int m = q >> 6;
            const int k = (q & 63) * 4;
            const float* gp = x + ((size_t)(b0 + m) * TSZ + t) * ISZ + k;
            const float4 v = *(const float4*)gp;
            sX[m * ISZ + k + 0] = (__bf16)v.x;
            sX[m * ISZ + k + 1] = (__bf16)v.y;
            sX[m * ISZ + k + 2] = (__bf16)v.z;
            sX[m * ISZ + k + 3] = (__bf16)v.w;
            if (t + 1 < TSZ) __builtin_prefetch(gp + ISZ, 0, 0);
        }
#endif
        __syncthreads();

        // ---- layer 0 gates: X(16x256)@Wih0^T + H0(16x64)@Whh0^T ----
        #pragma unroll
        for (int tt = 0; tt < 2; ++tt) {
            const int gbase = (wv * 2 + tt) * 16;
            v8f acc = {};
            #pragma unroll
            for (int kk = 0; kk < 8; ++kk) {        // K = 256
                v16bf a = load_frag_a(sX, ISZ, n, kk * 32, lane);
                v16bf b = load_frag_b(wih0bf, ISZ, gbase + n, kk * 32, lane);
                acc = __builtin_amdgcn_wmma_f32_16x16x32_bf16(
                        false, a, false, b, (short)0, acc, false, false);
            }
            #pragma unroll
            for (int kk = 0; kk < 2; ++kk) {        // K = 64 recurrent
                v16bf a = load_frag_a(sH0, HSZ, n, kk * 32, lane);
                v16bf b = load_frag_b(sWhh0, HSZ, gbase + n, kk * 32, lane);
                acc = __builtin_amdgcn_wmma_f32_16x16x32_bf16(
                        false, a, false, b, (short)0, acc, false, false);
            }
            #pragma unroll
            for (int r = 0; r < 8; ++r)
                sG[(r + moff) * GSZ + gbase + n] = acc[r];
        }
        __syncthreads();

        // ---- layer 0 elementwise; c in registers, h -> LDS bf16 ----
        #pragma unroll
        for (int k = 0; k < 4; ++k) {
            const int e = tid * 4 + k;
            const int m = e >> 6;
            const int j = e & 63;
            const float gi = sigmoidf_(sG[m * GSZ +           j] + sB0[j]);
            const float gf = sigmoidf_(sG[m * GSZ +  64 + j] + sB0[64 + j]);
            const float gg = tanhf_   (sG[m * GSZ + 128 + j] + sB0[128 + j]);
            const float go = sigmoidf_(sG[m * GSZ + 192 + j] + sB0[192 + j]);
            c0r[k] = gf * c0r[k] + gi * gg;
            sH0[m * HSZ + j] = (__bf16)(go * tanhf_(c0r[k]));
        }
        __syncthreads();

        // ---- layer 1 gates: H0(new)@Wih1^T + H1(old)@Whh1^T ----
        #pragma unroll
        for (int tt = 0; tt < 2; ++tt) {
            const int gbase = (wv * 2 + tt) * 16;
            v8f acc = {};
            #pragma unroll
            for (int kk = 0; kk < 2; ++kk) {
                v16bf a = load_frag_a(sH0, HSZ, n, kk * 32, lane);
                v16bf b = load_frag_b(sWih1, HSZ, gbase + n, kk * 32, lane);
                acc = __builtin_amdgcn_wmma_f32_16x16x32_bf16(
                        false, a, false, b, (short)0, acc, false, false);
            }
            #pragma unroll
            for (int kk = 0; kk < 2; ++kk) {
                v16bf a = load_frag_a(sH1, HSZ, n, kk * 32, lane);
                v16bf b = load_frag_b(sWhh1, HSZ, gbase + n, kk * 32, lane);
                acc = __builtin_amdgcn_wmma_f32_16x16x32_bf16(
                        false, a, false, b, (short)0, acc, false, false);
            }
            #pragma unroll
            for (int r = 0; r < 8; ++r)
                sG[(r + moff) * GSZ + gbase + n] = acc[r];
        }
        __syncthreads();

        // ---- layer 1 elementwise ----
        #pragma unroll
        for (int k = 0; k < 4; ++k) {
            const int e = tid * 4 + k;
            const int m = e >> 6;
            const int j = e & 63;
            const float gi = sigmoidf_(sG[m * GSZ +           j] + sB1[j]);
            const float gf = sigmoidf_(sG[m * GSZ +  64 + j] + sB1[64 + j]);
            const float gg = tanhf_   (sG[m * GSZ + 128 + j] + sB1[128 + j]);
            const float go = sigmoidf_(sG[m * GSZ + 192 + j] + sB1[192 + j]);
            c1r[k] = gf * c1r[k] + gi * gg;
            sH1[m * HSZ + j] = (__bf16)(go * tanhf_(c1r[k]));
        }
        __syncthreads();
    }

    // ---- FC head on final h1 ----
    if (tid < BTILE) {
        const int m = tid;
        float s = bfc[0];
        #pragma unroll 8
        for (int j = 0; j < HSZ; ++j)
            s += (float)sH1[m * HSZ + j] * wfc[j];
        out[b0 + m] = sigmoidf_(s);
    }
}

extern "C" void kernel_launch(void* const* d_in, const int* in_sizes, int n_in,
                              void* d_out, int out_size, void* d_ws, size_t ws_size,
                              hipStream_t stream) {
    const float* x    = (const float*)d_in[0];
    const float* Wih0 = (const float*)d_in[1];
    const float* Whh0 = (const float*)d_in[2];
    const float* bih0 = (const float*)d_in[3];
    const float* bhh0 = (const float*)d_in[4];
    const float* Wih1 = (const float*)d_in[5];
    const float* Whh1 = (const float*)d_in[6];
    const float* bih1 = (const float*)d_in[7];
    const float* bhh1 = (const float*)d_in[8];
    const float* Wfc  = (const float*)d_in[9];
    const float* bfc  = (const float*)d_in[10];
    float* out = (float*)d_out;

    // d_ws: bf16 copy of Wih0 (128 KB), L2-resident during the scan.
    __bf16* wih0bf = (__bf16*)d_ws;
    const int nW = GSZ * ISZ;
    convert_wih0_kernel<<<(nW + 255) / 256, 256, 0, stream>>>(Wih0, wih0bf, nW);

    lstm_fused_kernel<<<BSZ / BTILE, NTHREADS, 0, stream>>>(
        x, wih0bf, Whh0, bih0, bhh0, Wih1, Whh1, bih1, bhh1, Wfc, bfc, out);
}
